// Attention_32762010534254
// MI455X (gfx1250) — matrix operand order repaired and checked by reading, not compiled, not run on hardware
//
#include <hip/hip_runtime.h>

// ---------------------------------------------------------------------------
// Cross-attention for MI455X (gfx1250, wave32, WMMA + async global->LDS DMA).
// Pipeline: cvt weights->f16 | GEMM(q@Wq) | GEMM(kv@Wkv) | flash-attn | GEMM(@Wo+bo)
// Matmuls: v_wmma_f32_16x16x32_f16. Tile staging: global_load_async_to_lds_b128
// (ASYNCcnt) with double-buffered B tiles in the GEMMs.
// ---------------------------------------------------------------------------

typedef __attribute__((ext_vector_type(16))) _Float16 v16h;
typedef __attribute__((ext_vector_type(8)))  _Float16 v8h;
typedef __attribute__((ext_vector_type(4)))  _Float16 v4h;
typedef __attribute__((ext_vector_type(8)))  float    v8f;
typedef __attribute__((ext_vector_type(4)))  float    v4f;

#define WMMA(a, b, c) \
  __builtin_amdgcn_wmma_f32_16x16x32_f16(false, (a), false, (b), (short)0, (c), false, false)

__device__ inline v8f zero8() {
  v8f z = {0.f, 0.f, 0.f, 0.f, 0.f, 0.f, 0.f, 0.f};
  return z;
}

__device__ inline v16h pack2(v8h lo, v8h hi) {
  v16h r;
#pragma unroll
  for (int i = 0; i < 8; ++i) { r[i] = lo[i]; r[8 + i] = hi[i]; }
  return r;
}

// Async DMA: global -> LDS, 16B per lane, tracked by ASYNCcnt.
// INST_OFFSET is added to BOTH the LDS and global addresses (ISA 15.18.3).
template <int OFF>
__device__ __forceinline__ void async_b128(unsigned lds, const _Float16* g) {
  asm volatile("global_load_async_to_lds_b128 %0, %1, off offset:%2"
               :: "v"(lds), "v"(g), "i"(OFF) : "memory");
}
#define WAIT_ASYNC_0() asm volatile("s_wait_asynccnt 0x0" ::: "memory")
#define WAIT_ASYNC_2() asm volatile("s_wait_asynccnt 0x2" ::: "memory")
#define WAIT_DS_0()    asm volatile("s_wait_dscnt 0x0" ::: "memory")

// Flat LDS address low 32 bits == wave-relative LDS byte address.
__device__ __forceinline__ unsigned lds_addr(const void* p) {
  return (unsigned)(size_t)p;
}

// A-fragment (16x32, 16-bit) from LDS, row stride in halves (ISA 7.12.2).
__device__ inline v16h load_afrag_lds(const _Float16* base, int lane, int stride) {
  const int m  = lane & 15;
  const int kb = (lane >> 4) << 4;
  const _Float16* row = base + m * stride;
  v16h a;
#pragma unroll
  for (int i = 0; i < 8; ++i) a[i] = row[kb + i];
#pragma unroll
  for (int i = 0; i < 8; ++i) a[8 + i] = row[kb + 8 + i];
  return a;
}

// B-fragment (32x16, 16-bit) from LDS stored as [k][n], row stride in halves.
__device__ inline v16h load_bfrag_lds(const _Float16* base, int lane, int stride) {
  const int n  = lane & 15;
  const int kg = (lane >> 4) << 4;
  v16h b;
#pragma unroll
  for (int i = 0; i < 16; ++i) b[i] = base[(kg + i) * stride + n];
  return b;
}

// xor-reductions across the 16 lanes holding one C-matrix row, via ds_swizzle
// group-of-32 mode: offset = xor_mask<<10 | and_mask(0x1f).
template <int PAT>
__device__ __forceinline__ float swz(float v) {
  return __builtin_bit_cast(
      float, __builtin_amdgcn_ds_swizzle(__builtin_bit_cast(int, v), PAT));
}
__device__ inline float red_max16(float v) {
  float t;
  t = swz<0x041f>(v); v = v > t ? v : t;
  t = swz<0x081f>(v); v = v > t ? v : t;
  t = swz<0x101f>(v); v = v > t ? v : t;
  t = swz<0x201f>(v); v = v > t ? v : t;
  return v;
}
__device__ inline float red_sum16(float v) {
  v += swz<0x041f>(v);
  v += swz<0x081f>(v);
  v += swz<0x101f>(v);
  v += swz<0x201f>(v);
  return v;
}

// ---------------------------------------------------------------------------
// f32 -> f16 elementwise (weights), 4 elements/thread.
// ---------------------------------------------------------------------------
__global__ __launch_bounds__(256) void cvt_f32_f16(const float* __restrict__ s,
                                                   _Float16* __restrict__ d,
                                                   int n4) {
  const int i = blockIdx.x * 256 + threadIdx.x;
  if (i < n4) {
    const v4f x = *(const v4f*)(s + (size_t)i * 4);
    v4h y;
#pragma unroll
    for (int t = 0; t < 4; ++t) y[t] = (_Float16)x[t];
    *(v4h*)(d + (size_t)i * 4) = y;
  }
}

// ---------------------------------------------------------------------------
// 128x128-tile WMMA GEMM: C[M,N] = A[M,K] * B16[K,N] (+bias).
// A f32 (converted while staging) or f16 (async-DMA staged). B f16, staged by
// global_load_async_to_lds_b128 into double-buffered LDS tiles.
// 256 threads = 8 waves; each wave computes 32x64.
// ---------------------------------------------------------------------------
template <typename TA>
__global__ __launch_bounds__(256) void gemm_wmma_128x128(
    const TA* __restrict__ A, int lda, const _Float16* __restrict__ B, int ldb,
    _Float16* __restrict__ Ch, float* __restrict__ Cf,
    const float* __restrict__ bias, int ldc, int K) {
  __shared__ alignas(16) _Float16 As[128][40];      // 128x32 tile (padded)
  __shared__ alignas(16) _Float16 Bs[2][32][136];   // 2 x (32x128) tiles

  const int tid  = threadIdx.x;
  const int lane = tid & 31;
  const int w    = tid >> 5;
  const int wm   = w >> 1;  // 0..3
  const int wn   = w & 1;   // 0..1
  const int bm   = blockIdx.y * 128;
  const int bn   = blockIdx.x * 128;

  // Async B staging: wave w covers rows 4w..4w+3; each row = 256B = 16 chunks.
  const int brow   = w * 4 + (lane >> 3);
  const int bchunk = lane & 7;  // chunks 0..7; second instr adds +8 (offset 128)
  auto issueB = [&](int kc, int buf) {
    const unsigned l = lds_addr(&Bs[buf][brow][bchunk * 8]);
    const _Float16* g = B + (size_t)(kc + brow) * ldb + bn + bchunk * 8;
    async_b128<0>(l, g);
    async_b128<128>(l, g);
  };

  auto stageA = [&](int kc) {
    if constexpr (sizeof(TA) == 4) {
      // f32 -> f16 convert while staging (4 elems/thread x 4).
#pragma unroll
      for (int j = 0; j < 4; ++j) {
        const int e   = tid + j * 256;
        const int row = e >> 3;
        const int col = (e & 7) << 2;
        const float* src = (const float*)A + (size_t)(bm + row) * lda + kc + col;
        const v4f x = *(const v4f*)src;
        v4h y;
#pragma unroll
        for (int t = 0; t < 4; ++t) y[t] = (_Float16)x[t];
        *(v4h*)&As[row][col] = y;
      }
    } else {
      // f16: async DMA. Wave w covers rows 16w..16w+15; row = 64B = 4 chunks.
      const int arow = w * 16 + (lane >> 1);
      const int acol = (lane & 1) * 8;  // halves
      const unsigned l = lds_addr(&As[arow][acol]);
      const _Float16* g = (const _Float16*)A + (size_t)(bm + arow) * lda + kc + acol;
      async_b128<0>(l, g);
      async_b128<32>(l, g);
    }
  };

  v8f acc[2][4];
#pragma unroll
  for (int mt = 0; mt < 2; ++mt)
#pragma unroll
    for (int nt = 0; nt < 4; ++nt) acc[mt][nt] = zero8();

  int buf = 0;
  issueB(0, 0);
  for (int kc = 0; kc < K; kc += 32) {
    stageA(kc);
    const bool more = (kc + 32) < K;
    if (more) {
      issueB(kc + 32, buf ^ 1);
      WAIT_ASYNC_2();  // in-order: retires current-buffer (and A) transfers
    } else {
      WAIT_ASYNC_0();
    }
    __syncthreads();

    v16h af0 = load_afrag_lds(&As[wm * 32][0], lane, 40);
    v16h af1 = load_afrag_lds(&As[wm * 32 + 16][0], lane, 40);
#pragma unroll
    for (int nt = 0; nt < 4; ++nt) {
      v16h bf = load_bfrag_lds(&Bs[buf][0][wn * 64 + nt * 16], lane, 136);
      acc[0][nt] = WMMA(af0, bf, acc[0][nt]);
      acc[1][nt] = WMMA(af1, bf, acc[1][nt]);
    }
    __syncthreads();
    buf ^= 1;
  }

  const int rbase = (lane >> 4) << 3;
  const int n     = lane & 15;
#pragma unroll
  for (int mt = 0; mt < 2; ++mt)
#pragma unroll
    for (int nt = 0; nt < 4; ++nt)
#pragma unroll
      for (int r = 0; r < 8; ++r) {
        const int row = bm + wm * 32 + mt * 16 + rbase + r;
        const int col = bn + wn * 64 + nt * 16 + n;
        if (Cf != nullptr)
          Cf[(size_t)row * ldc + col] = acc[mt][nt][r] + bias[col];
        else
          Ch[(size_t)row * ldc + col] = (_Float16)acc[mt][nt][r];
      }
}

// ---------------------------------------------------------------------------
// Flash attention: one wave per 16-row Q tile per (b, head).
// Qp: (B*Nq, 1024) f16, col = h*64+d. KVp: (B*Nkv, 2048) f16, V at +1024.
// kv blocks of 32: S=Q*K^T (K^T frags contiguous along dh), online softmax,
// O += P*V with V async-DMA'd to LDS and P re-laid-out through LDS.
// ---------------------------------------------------------------------------
#define A_B    2
#define A_NQ   512
#define A_NKV  4096
#define A_DH   64
#define A_SCL  0.125f  // 1/sqrt(64)

__global__ __launch_bounds__(128) void attn_wmma(
    const _Float16* __restrict__ Qp, const _Float16* __restrict__ KVp,
    const unsigned char* __restrict__ mask, _Float16* __restrict__ O) {
  __shared__ alignas(16) _Float16 Vs[4][32][72];  // per-wave 32kv x 64dh
  __shared__ alignas(16) _Float16 Ps[4][16][40];  // per-wave 16q x 32kv

  const int lane = threadIdx.x & 31;
  const int wave = threadIdx.x >> 5;
  const int task = blockIdx.x * 4 + wave;  // 0..1023
  const int qt   = task & 31;
  const int bh   = task >> 5;
  const int h    = bh & 15;
  const int b    = bh >> 4;
  const int qbase = qt * 16;

  const int m     = lane & 15;
  const int kb    = (lane >> 4) << 4;  // A-frag K base
  const int n     = lane & 15;
  const int kg    = (lane >> 4) << 4;  // B-frag K group
  const int rbase = (lane >> 4) << 3;  // C-frag row base

  const _Float16* qrow = Qp + (size_t)(b * A_NQ + qbase + m) * 1024 + h * A_DH;
  v16h qf[2];
#pragma unroll
  for (int c = 0; c < 2; ++c)
    qf[c] = pack2(*(const v8h*)(qrow + c * 32 + kb),
                  *(const v8h*)(qrow + c * 32 + kb + 8));

  v8f o[4];
#pragma unroll
  for (int nt = 0; nt < 4; ++nt) o[nt] = zero8();
  float rm[8], rs[8];
#pragma unroll
  for (int r = 0; r < 8; ++r) { rm[r] = -1e30f; rs[r] = 0.f; }

  const unsigned vlds = lds_addr(&Vs[wave][lane][0]);

  for (int kv0 = 0; kv0 < A_NKV; kv0 += 32) {
    // Kick off V DMA for this block first so it hides under S + softmax.
    // Prior iteration's LDS reads must be retired before the overwrite.
    WAIT_DS_0();
    {
      const _Float16* g =
          KVp + (size_t)(b * A_NKV + kv0 + lane) * 2048 + 1024 + h * A_DH;
      async_b128<0>(vlds, g);   async_b128<16>(vlds, g);
      async_b128<32>(vlds, g);  async_b128<48>(vlds, g);
      async_b128<64>(vlds, g);  async_b128<80>(vlds, g);
      async_b128<96>(vlds, g);  async_b128<112>(vlds, g);
    }

    // S = Q * K^T.
    v8f s[2];
#pragma unroll
    for (int kvt = 0; kvt < 2; ++kvt) {
      const _Float16* krow =
          KVp + (size_t)(b * A_NKV + kv0 + kvt * 16 + n) * 2048 + h * A_DH;
      v8f c = zero8();
#pragma unroll
      for (int cc = 0; cc < 2; ++cc) {
        v16h bf = pack2(*(const v8h*)(krow + cc * 32 + kg),
                        *(const v8h*)(krow + cc * 32 + kg + 8));
        c = WMMA(qf[cc], bf, c);
      }
      s[kvt] = c;
    }

    const float madd0 = mask[b * A_NKV + kv0 + n] ? 0.f : -1e30f;
    const float madd1 = mask[b * A_NKV + kv0 + 16 + n] ? 0.f : -1e30f;
    float p0[8], p1[8];
#pragma unroll
    for (int r = 0; r < 8; ++r) {
      p0[r] = s[0][r] * A_SCL + madd0;
      p1[r] = s[1][r] * A_SCL + madd1;
    }

    // Online softmax (rows live across 16 lanes of each half-wave).
#pragma unroll
    for (int r = 0; r < 8; ++r) {
      const float pmx  = p0[r] > p1[r] ? p0[r] : p1[r];
      const float bmax = red_max16(pmx);
      const float mn   = rm[r] > bmax ? rm[r] : bmax;
      const float corr = __expf(rm[r] - mn);
      rm[r] = mn;
      p0[r] = __expf(p0[r] - mn);
      p1[r] = __expf(p1[r] - mn);
      const float rsum = red_sum16(p0[r] + p1[r]);
      rs[r] = rs[r] * corr + rsum;
#pragma unroll
      for (int nt = 0; nt < 4; ++nt) o[nt][r] *= corr;
      Ps[wave][rbase + r][n]      = (_Float16)p0[r];
      Ps[wave][rbase + r][16 + n] = (_Float16)p1[r];
    }

    // P stores (DS, in-order per wave) and V DMA must land before frag reads.
    WAIT_DS_0();
    WAIT_ASYNC_0();

    v16h pa = load_afrag_lds(&Ps[wave][0][0], lane, 40);
#pragma unroll
    for (int nt = 0; nt < 4; ++nt) {
      v16h vb = load_bfrag_lds(&Vs[wave][0][nt * 16], lane, 72);
      o[nt] = WMMA(pa, vb, o[nt]);
    }
  }

#pragma unroll
  for (int nt = 0; nt < 4; ++nt)
#pragma unroll
    for (int r = 0; r < 8; ++r) {
      const float v = o[nt][r] / rs[r];
      O[(size_t)(b * A_NQ + qbase + rbase + r) * 1024 + h * A_DH + nt * 16 + n] =
          (_Float16)v;
    }
}

// ---------------------------------------------------------------------------
// Host launcher. Inputs: q, kv, mask(bool,1B), Wq, Wkv, Wo, bo (f32).
// Workspace (f16): Qp 1M | KVp 16M | AttnOut 1M | Wq16 1M | Wkv16 2M | Wo16 1M
// halves  (~46 MB total).
// ---------------------------------------------------------------------------
extern "C" void kernel_launch(void* const* d_in, const int* in_sizes, int n_in,
                              void* d_out, int out_size, void* d_ws,
                              size_t ws_size, hipStream_t stream) {
  (void)in_sizes; (void)n_in; (void)out_size; (void)ws_size;

  const float* q    = (const float*)d_in[0];
  const float* kv   = (const float*)d_in[1];
  const unsigned char* mask = (const unsigned char*)d_in[2];
  const float* Wq   = (const float*)d_in[3];
  const float* Wkv  = (const float*)d_in[4];
  const float* Wo   = (const float*)d_in[5];
  const float* bo   = (const float*)d_in[6];
  float* out = (float*)d_out;

  _Float16* Qp      = (_Float16*)d_ws;                     // 1024x1024
  _Float16* KVp     = Qp + (size_t)1024 * 1024;            // 8192x2048
  _Float16* AttnOut = KVp + (size_t)8192 * 2048;           // 1024x1024
  _Float16* Wq16    = AttnOut + (size_t)1024 * 1024;       // 1024x1024
  _Float16* Wkv16   = Wq16 + (size_t)1024 * 1024;          // 1024x2048
  _Float16* Wo16    = Wkv16 + (size_t)1024 * 2048;         // 1024x1024

  // Weights -> f16 (enables verbatim async-DMA staging in the GEMMs).
  cvt_f32_f16<<<dim3(1024 * 1024 / 1024), 256, 0, stream>>>(Wq, Wq16,
                                                            1024 * 1024 / 4);
  cvt_f32_f16<<<dim3(2 * 1024 * 1024 / 1024), 256, 0, stream>>>(
      Wkv, Wkv16, 2 * 1024 * 1024 / 4);
  cvt_f32_f16<<<dim3(1024 * 1024 / 1024), 256, 0, stream>>>(Wo, Wo16,
                                                            1024 * 1024 / 4);

  // Q projection: (1024 x 1024) = q @ Wq
  gemm_wmma_128x128<float><<<dim3(8, 8), 256, 0, stream>>>(
      q, 1024, Wq16, 1024, Qp, nullptr, nullptr, 1024, 1024);

  // KV projection: (8192 x 2048) = kv @ Wkv
  gemm_wmma_128x128<float><<<dim3(16, 64), 256, 0, stream>>>(
      kv, 1024, Wkv16, 2048, KVp, nullptr, nullptr, 2048, 1024);

  // Attention: 1024 wave-tasks, 4 waves/block.
  attn_wmma<<<dim3(256), 128, 0, stream>>>(Qp, KVp, mask, AttnOut);

  // Output projection with bias: (1024 x 1024) f32 out, f16 A async-staged.
  gemm_wmma_128x128<_Float16><<<dim3(8, 8), 256, 0, stream>>>(
      AttnOut, 1024, Wo16, 1024, nullptr, out, bo, 1024, 1024);
}
